// Net_2190433321475
// MI455X (gfx1250) — matrix operand order repaired and checked by reading, not compiled, not run on hardware
//
#include <hip/hip_runtime.h>
#include <hip/hip_bf16.h>
#include <math.h>

typedef _Float16 v16h __attribute__((ext_vector_type(16)));
typedef _Float16 v8h  __attribute__((ext_vector_type(8)));
typedef float    v8f  __attribute__((ext_vector_type(8)));

#define B_   32
#define T_   2000
#define W_   512
#define G_   20
#define S_   100
#define H_   1024
#define O_   10
#define BETA 0.95f

// ---------------------------------------------------------------------------
// Kernel 1: fused 1x1 conv (C=4) + 100-step leaky scan. One thread = one
// (b,g,w) sequence. Streams all 524 MB of x exactly once, coalesced float4.
// Emits final spike of scan #1 as f16 (exact: spike in {0,1}) for WMMA A.
// ---------------------------------------------------------------------------
__global__ __launch_bounds__(256) void k_fuse_scan1(
    const float4* __restrict__ x,       // [B*T*W] float4 over C
    const float*  __restrict__ conv_w,  // [4]
    const float*  __restrict__ conv_b,  // [1]
    _Float16*     __restrict__ spk1)    // [640, 512]  row n = b*G+g
{
  int tid = blockIdx.x * blockDim.x + threadIdx.x;   // 0 .. 327679
  int w = tid & (W_ - 1);
  int n = tid >> 9;                                   // b*G + g
  int b = n / G_;
  int g = n - b * G_;

  const float4 cw = *(const float4*)conv_w;
  const float  cb = conv_b[0];

  int base = (b * T_ + g * S_) * W_ + w;              // float4 index
  float mem = 0.f;
#pragma unroll 4
  for (int s = 0; s < S_; ++s) {
    float4 v = x[base + s * W_];
    float xt = v.x * cw.x + v.y * cw.y + v.z * cw.z + v.w * cw.w + cb;
    float nb = BETA * mem + xt;
    mem = (mem > 1.0f) ? 0.f : nb;                    // zero-reset when prev mem spiked
  }
  spk1[tid] = (_Float16)((mem > 1.0f) ? 1.0f : 0.0f);
}

// ---------------------------------------------------------------------------
// Kernel 2: split fp32 hidden weights into hi/lo f16 halves (w = hi + lo)
// so the f16 WMMA pair reproduces fp32-accurate products.
// ---------------------------------------------------------------------------
__global__ __launch_bounds__(256) void k_split_w(
    const float* __restrict__ w, _Float16* __restrict__ hi,
    _Float16* __restrict__ lo, int nElem)
{
  int i = blockIdx.x * blockDim.x + threadIdx.x;
  if (i < nElem) {
    float v = w[i];
    _Float16 h = (_Float16)v;
    hi[i] = h;
    lo[i] = (_Float16)(v - (float)h);
  }
}

// ---------------------------------------------------------------------------
// Kernel 3: hidden GEMM via v_wmma_f32_16x16x32_f16.
// D[640,1024] = A[640,512] * W^T + bias, with W split as Whi + Wlo.
// One 16x16 tile per wave; K-loop: 512/32 = 16 steps, 2 WMMAs per step.
// Operand k order: contiguous k = 16*(lane/16) + s in both A and B (wave32).
// ---------------------------------------------------------------------------
__global__ __launch_bounds__(256) void k_gemm_hidden(
    const _Float16* __restrict__ A,    // [640, 512] spikes
    const _Float16* __restrict__ Bhi,  // [1024, 512]
    const _Float16* __restrict__ Blo,  // [1024, 512]
    const float*    __restrict__ bias, // [1024]
    float*          __restrict__ C)    // [640, 1024]
{
  int wave = (blockIdx.x * blockDim.x + threadIdx.x) >> 5;   // 0 .. 2559
  int lane = threadIdx.x & 31;
  int tm = wave >> 6;          // 0 .. 39  (M tiles)
  int tn = wave & 63;          // 0 .. 63  (N tiles)
  int m0 = tm * 16, n0 = tn * 16;

  int hl   = lane & 15;
  int hiln = lane >> 4;        // 0: lanes 0-15, 1: lanes 16-31

  // A: lane holds row m = m0+hl; kbase 0 (lo lanes) / 16 (hi lanes);
  //    half j -> K = kA+j (contiguous)
  const _Float16* Arow = A + (m0 + hl) * W_ + (hiln ? 16 : 0);
  // B: lane holds col n = n0+hl; kbase 0 / 16; half j -> K = kB+j (contiguous)
  const _Float16* Bhrow = Bhi + (n0 + hl) * W_ + (hiln ? 16 : 0);
  const _Float16* Blrow = Blo + (n0 + hl) * W_ + (hiln ? 16 : 0);

  v8f acc = {0.f, 0.f, 0.f, 0.f, 0.f, 0.f, 0.f, 0.f};

#pragma unroll 4
  for (int kk = 0; kk < W_ / 32; ++kk) {
    int k = kk * 32;
    v8h a_lo = *(const v8h*)(Arow + k);        // K = kA+0 .. kA+7
    v8h a_hi = *(const v8h*)(Arow + k + 8);    // K = kA+8 .. kA+15
    v16h a;
#pragma unroll
    for (int j = 0; j < 8; ++j) { a[j] = a_lo[j]; a[8 + j] = a_hi[j]; }

    v16h bh = *(const v16h*)(Bhrow + k);       // 16 contiguous K per lane
    v16h bl = *(const v16h*)(Blrow + k);

    acc = __builtin_amdgcn_wmma_f32_16x16x32_f16(false, a, false, bh,
                                                 (short)0, acc, false, false);
    acc = __builtin_amdgcn_wmma_f32_16x16x32_f16(false, a, false, bl,
                                                 (short)0, acc, false, false);
  }

  // D layout: lane L, reg r -> M = r + (L>=16 ? 8 : 0), N = L&15
  float bval = bias[n0 + hl];
  int mbase = m0 + (hiln ? 8 : 0);
#pragma unroll
  for (int r = 0; r < 8; ++r) {
    C[(mbase + r) * H_ + n0 + hl] = acc[r] + bval;
  }
}

// ---------------------------------------------------------------------------
// Kernel 4: second leaky scan over G=20 groups. One thread per (b,h).
// ---------------------------------------------------------------------------
__global__ __launch_bounds__(256) void k_scan2(
    const float* __restrict__ hidden,  // [640, 1024] rows n = b*G+g
    float*       __restrict__ spk2)    // [32, 1024]
{
  int tid = blockIdx.x * blockDim.x + threadIdx.x;  // 0 .. 32767
  int b = tid >> 10;
  int h = tid & (H_ - 1);
  float mem = 0.f;
#pragma unroll
  for (int g = 0; g < G_; ++g) {
    float xt = hidden[(b * G_ + g) * H_ + h];
    float nb = BETA * mem + xt;
    mem = (mem > 1.0f) ? 0.f : nb;
  }
  spk2[tid] = (mem > 1.0f) ? 1.0f : 0.0f;
}

// ---------------------------------------------------------------------------
// Kernel 5: output layer + softmax. One block per batch element.
// ---------------------------------------------------------------------------
__global__ __launch_bounds__(256) void k_out_softmax(
    const float* __restrict__ spk2,   // [32, 1024]
    const float* __restrict__ out_w,  // [10, 1024]
    const float* __restrict__ out_b,  // [10]
    float*       __restrict__ out)    // [32, 10]
{
  __shared__ float red[256];
  __shared__ float logits[O_];
  int b = blockIdx.x;
  int t = threadIdx.x;

  float acc[O_];
#pragma unroll
  for (int o = 0; o < O_; ++o) acc[o] = 0.f;

  for (int h = t; h < H_; h += 256) {
    float s = spk2[b * H_ + h];
#pragma unroll
    for (int o = 0; o < O_; ++o) acc[o] += s * out_w[o * H_ + h];
  }

  for (int o = 0; o < O_; ++o) {
    red[t] = acc[o];
    __syncthreads();
    for (int off = 128; off > 0; off >>= 1) {
      if (t < off) red[t] += red[t + off];
      __syncthreads();
    }
    if (t == 0) logits[o] = red[0] + out_b[o];
    __syncthreads();
  }

  if (t == 0) {
    float mx = logits[0];
#pragma unroll
    for (int o = 1; o < O_; ++o) mx = fmaxf(mx, logits[o]);
    float e[O_], sum = 0.f;
#pragma unroll
    for (int o = 0; o < O_; ++o) { e[o] = expf(logits[o] - mx); sum += e[o]; }
    float inv = 1.0f / sum;
#pragma unroll
    for (int o = 0; o < O_; ++o) out[b * O_ + o] = e[o] * inv;
  }
}

// ---------------------------------------------------------------------------
extern "C" void kernel_launch(void* const* d_in, const int* in_sizes, int n_in,
                              void* d_out, int out_size, void* d_ws, size_t ws_size,
                              hipStream_t stream) {
  const float* x        = (const float*)d_in[0];  // [32,2000,512,4]
  const float* conv_w   = (const float*)d_in[1];  // [4]
  const float* conv_b   = (const float*)d_in[2];  // [1]
  const float* hidden_w = (const float*)d_in[3];  // [1024,512]
  const float* hidden_b = (const float*)d_in[4];  // [1024]
  const float* out_w    = (const float*)d_in[5];  // [10,1024]
  const float* out_b    = (const float*)d_in[6];  // [10]
  float* out = (float*)d_out;                     // [32,10]

  char* ws = (char*)d_ws;
  // workspace layout (bytes)
  _Float16* spk1 = (_Float16*)(ws + 0);          // 640*512*2   = 655360
  _Float16* whi  = (_Float16*)(ws + 655360);     // 1024*512*2  = 1048576
  _Float16* wlo  = (_Float16*)(ws + 1703936);    // 1024*512*2  = 1048576
  float*  hidden = (float*)   (ws + 2752512);    // 640*1024*4  = 2621440
  float*  spk2   = (float*)   (ws + 5373952);    // 32*1024*4   = 131072

  // 1) conv fuse + scan1: 327,680 threads (memory-bound, 524 MB stream)
  k_fuse_scan1<<<(B_ * G_ * W_) / 256, 256, 0, stream>>>(
      (const float4*)x, conv_w, conv_b, spk1);

  // 2) split hidden weights into f16 hi/lo
  k_split_w<<<(H_ * W_) / 256, 256, 0, stream>>>(hidden_w, whi, wlo, H_ * W_);

  // 3) WMMA GEMM: 2560 tiles, 8 waves/block -> 320 blocks
  k_gemm_hidden<<<320, 256, 0, stream>>>(spk1, whi, wlo, hidden_b, hidden);

  // 4) scan2: 32768 threads
  k_scan2<<<(B_ * H_) / 256, 256, 0, stream>>>(hidden, spk2);

  // 5) output layer + softmax
  k_out_softmax<<<B_, 256, 0, stream>>>(spk2, out_w, out_b, out);
}
